// ContrastiveLoss_20169166422820
// MI455X (gfx1250) — matrix operand order repaired and checked by reading, not compiled, not run on hardware
//
#include <hip/hip_runtime.h>

// ---------------- problem constants ----------------
#define N_ROWS 4096
#define DDIM   768
#define INV_T  20.0f          // 1 / 0.05

// ---------------- tiling ----------------
#define JB      128                    // anchor rows per workgroup (8 waves x 16)
#define STRIPES 16                     // column stripes across the grid
#define SW      (N_ROWS / STRIPES)     // 256 columns per stripe
#define IT_W    32                     // columns staged/computed per iteration (2 subtiles)
#define ITILES  (SW / IT_W)            // 8 iterations per stripe
#define KSTEPS  (DDIM / 32)            // 24 WMMA K-steps
#define A_RS    1552                   // LDS row stride bytes: 768*2 + 16 pad (bank-friendly)

typedef __bf16 bf16;
typedef __attribute__((ext_vector_type(16))) __bf16 v16bf;
typedef __attribute__((ext_vector_type(8)))  float  v8f;

union Frag { uint4 u[2]; v16bf v; };
struct __align__(8) BH4 { bf16 x, y, z, w; };

// ============================================================
// Phase 1: tiled bf16 WMMA GEMM + fused exp row-sum epilogue
//   Each wave computes a 16(M) x 32(N) tile: one A fragment feeds two WMMAs.
//   wsDen[row*STRIPES + stripe] = sum_{i in stripe} exp(logit) (+diag twice)
//   wsLast[row]                 = logits[row, N-1]
// ============================================================
__global__ __launch_bounds__(256)
void contrastive_gemm_kernel(const float* __restrict__ anchors,
                             const float* __restrict__ positives,
                             float* __restrict__ wsDen,
                             float* __restrict__ wsLast) {
  extern __shared__ char smem[];
  char* As = smem;                        // JB rows   x A_RS bytes (bf16 anchors)
  char* Bs = smem + (size_t)JB * A_RS;    // IT_W rows x A_RS bytes (bf16 positives tile)

  const int t      = threadIdx.x;
  const int stripe = blockIdx.x;
  const int j0     = blockIdx.y * JB;
  const int iBase  = stripe * SW;

  // ---- stage A: anchors[j0 : j0+128, :] -> bf16 LDS (row-major) ----
  #pragma unroll 4
  for (int q = 0; q < (JB * DDIM / 4) / 256; ++q) {   // 96 float4 per thread
    int f   = t + 256 * q;
    int row = f / (DDIM / 4);
    int c4  = f % (DDIM / 4);
    float4 x = ((const float4*)(anchors + (size_t)(j0 + row) * DDIM))[c4];
    BH4 h = { (bf16)x.x, (bf16)x.y, (bf16)x.z, (bf16)x.w };
    *(BH4*)(As + row * A_RS + c4 * 8) = h;
  }

  const int wv   = t >> 5;          // wave id 0..7  -> 16-row sub-block
  const int lane = t & 31;
  const int half = lane >> 4;       // 0: lanes 0-15, 1: lanes 16-31
  const int l16  = lane & 15;
  const int rowG = j0 + wv * 16 + half * 8;   // + r gives global row of C elem r

  // A 16x32 layout (matches B): lanes 0-15 hold K 0..15, lanes 16-31 hold K 16..31
  const char* Abase  = As + (wv * 16 + l16) * A_RS + half * 32;
  // B 32x16 layout (matches A): lanes 0-15 hold K 0..15 (col=lane), lanes 16-31 hold K 16..31
  const char* Bbase0 = Bs + l16 * A_RS + half * 32;
  const char* Bbase1 = Bbase0 + 16 * A_RS;

  float acc[8];
  #pragma unroll
  for (int r = 0; r < 8; ++r) acc[r] = 0.0f;

  for (int it = 0; it < ITILES; ++it) {
    const int i0 = iBase + it * IT_W;
    __syncthreads();   // A ready (it==0) / previous tile's B readers done

    // ---- stage B tile: positives[i0 : i0+32, :] -> bf16 LDS ----
    #pragma unroll 4
    for (int q = 0; q < (IT_W * DDIM / 4) / 256; ++q) {  // 24 float4 per thread
      int f   = t + 256 * q;
      int row = f / (DDIM / 4);
      int c4  = f % (DDIM / 4);
      float4 x = ((const float4*)(positives + (size_t)(i0 + row) * DDIM))[c4];
      BH4 h = { (bf16)x.x, (bf16)x.y, (bf16)x.z, (bf16)x.w };
      *(BH4*)(Bs + row * A_RS + c4 * 8) = h;
    }
    __syncthreads();

    // ---- K loop: 2-deep software pipeline, one A frag feeds two WMMAs ----
    v8f c0 = {0.f,0.f,0.f,0.f,0.f,0.f,0.f,0.f};
    v8f c1 = {0.f,0.f,0.f,0.f,0.f,0.f,0.f,0.f};
    Frag a[2], b0[2], b1[2];

    // prologue: fragments for kk = 0
    a [0].u[0] = *(const uint4*)(Abase);
    a [0].u[1] = *(const uint4*)(Abase + 16);
    b0[0].u[0] = *(const uint4*)(Bbase0);
    b0[0].u[1] = *(const uint4*)(Bbase0 + 16);
    b1[0].u[0] = *(const uint4*)(Bbase1);
    b1[0].u[1] = *(const uint4*)(Bbase1 + 16);

    #pragma unroll
    for (int kk = 0; kk < KSTEPS; ++kk) {
      const int cur = kk & 1;
      const int nxt = cur ^ 1;
      if (kk < KSTEPS - 1) {                 // issue next-step loads before compute
        const int ko = (kk + 1) * 64;
        a [nxt].u[0] = *(const uint4*)(Abase  + ko);
        a [nxt].u[1] = *(const uint4*)(Abase  + ko + 16);
        b0[nxt].u[0] = *(const uint4*)(Bbase0 + ko);
        b0[nxt].u[1] = *(const uint4*)(Bbase0 + ko + 16);
        b1[nxt].u[0] = *(const uint4*)(Bbase1 + ko);
        b1[nxt].u[1] = *(const uint4*)(Bbase1 + ko + 16);
      }
      c0 = __builtin_amdgcn_wmma_f32_16x16x32_bf16(false, a[cur].v, false, b0[cur].v,
                                                   (short)0, c0, false, false);
      c1 = __builtin_amdgcn_wmma_f32_16x16x32_bf16(false, a[cur].v, false, b1[cur].v,
                                                   (short)0, c1, false, false);
    }

    // ---- fused epilogue: scale by 1/T, exp, diagonal double-count, last col ----
    const int colG0 = i0 + l16;
    const int colG1 = i0 + 16 + l16;
    #pragma unroll
    for (int r = 0; r < 8; ++r) {
      int   rG = rowG + r;
      float l0 = c0[r] * INV_T;
      float l1 = c1[r] * INV_T;
      float e0 = __expf(l0);
      float e1 = __expf(l1);
      acc[r] += (colG0 == rG) ? 2.0f * e0 : e0;    // denom counts diag twice
      acc[r] += (colG1 == rG) ? 2.0f * e1 : e1;
      // column N-1 is congruent to 31 mod 32 -> always lives in subtile 1
      if (colG1 == N_ROWS - 1) wsLast[rG] = l1;    // logits[:, -1]
    }
  }

  // ---- row-sum across the 16 lanes of each half-wave (C layout: N = lane&15) ----
  #pragma unroll
  for (int r = 0; r < 8; ++r) {
    float v = acc[r];
    v += __shfl_xor(v, 8, 16);
    v += __shfl_xor(v, 4, 16);
    v += __shfl_xor(v, 2, 16);
    v += __shfl_xor(v, 1, 16);
    if (l16 == 0)
      wsDen[(size_t)(rowG + r) * STRIPES + stripe] = v;
  }
}

// ============================================================
// Phase 2: deterministic single-block reduction to the scalar loss
// ============================================================
__global__ __launch_bounds__(1024)
void contrastive_finalize_kernel(const float* __restrict__ wsDen,
                                 const float* __restrict__ wsLast,
                                 float* __restrict__ out) {
  __shared__ float red[1024];
  const int t = threadIdx.x;
  float s = 0.f;
  for (int r = t; r < N_ROWS; r += 1024) {
    float d = 0.f;
    #pragma unroll
    for (int q = 0; q < STRIPES; ++q) d += wsDen[(size_t)r * STRIPES + q];
    s += __logf(d) - wsLast[r];
  }
  red[t] = s;
  __syncthreads();
  for (int off = 512; off > 0; off >>= 1) {
    if (t < off) red[t] += red[t + off];
    __syncthreads();
  }
  if (t == 0) out[0] = red[0];
}

// ============================================================
extern "C" void kernel_launch(void* const* d_in, const int* in_sizes, int n_in,
                              void* d_out, int out_size, void* d_ws, size_t ws_size,
                              hipStream_t stream) {
  (void)in_sizes; (void)n_in; (void)out_size; (void)ws_size;
  const float* anchors   = (const float*)d_in[0];
  const float* positives = (const float*)d_in[1];
  float* wsDen  = (float*)d_ws;                       // N_ROWS * STRIPES floats
  float* wsLast = wsDen + (size_t)N_ROWS * STRIPES;   // N_ROWS floats

  const size_t lds = (size_t)JB * A_RS + (size_t)IT_W * A_RS;   // ~248 KB
  dim3 grid(STRIPES, N_ROWS / JB);                    // 16 x 32 = 512 workgroups
  contrastive_gemm_kernel<<<grid, 256, lds, stream>>>(anchors, positives, wsDen, wsLast);
  contrastive_finalize_kernel<<<1, 1024, 0, stream>>>(wsDen, wsLast, (float*)d_out);
}
